// MultiScaleAttention_38122129719490
// MI455X (gfx1250) — matrix operand order repaired and checked by reading, not compiled, not run on hardware
//
#include <hip/hip_runtime.h>

// ---------------------------------------------------------------------------
// MultiScaleAttention for MI455X (gfx1250)
//   - all GEMMs via v_wmma_f32_16x16x32_bf16 (f32 accumulate)
//   - K2 uses double-buffered global_load_async_to_lds_b128 + s_wait_asynccnt
//   - producers emit column-major bf16 copies so K4/K5 B-fragments are direct
// ---------------------------------------------------------------------------

typedef __attribute__((ext_vector_type(16))) __bf16 v16bf;
typedef __attribute__((ext_vector_type(8)))  __bf16 v8bf;
typedef __attribute__((ext_vector_type(8)))  float  v8f;

#define DEV __device__ __forceinline__

constexpr int B_  = 8;
constexpr int C_  = 256;     // input/output channels
constexpr int CC_ = 128;     // attention channels
constexpr int HW_ = 16384;   // 128*128
constexpr int NT1 = 32;      // hw tile per block for K1

DEV v8f wmma_bf16(v16bf a, v16bf b, v8f c) {
    // D = A(16x32) * B(32x16) + C, f32 accumulate
    return __builtin_amdgcn_wmma_f32_16x16x32_bf16(
        /*neg_a=*/false, a, /*neg_b=*/false, b,
        /*c_mod=*/(short)0, c, /*reuse_a=*/false, /*reuse_b=*/false);
}

// A fragment (16x32 bf16, MxK), source row-major [rows][ld] in bf16.
// ISA: lanes 0-15 hold M=lane, K={0..7,16..23}; lanes 16-31: K={8..15,24..31}
DEV v16bf load_a_bf16(const __bf16* base, int ld, int m0, int k0, int lane) {
    int r  = m0 + (lane & 15);
    int kb = k0 + ((lane >> 4) << 3);
    const __bf16* p = base + (size_t)r * ld + kb;
    union { v16bf v; v8bf h[2]; } u;
    u.h[0] = *(const v8bf*)(p);
    u.h[1] = *(const v8bf*)(p + 16);
    return u.v;
}

// Same A fragment, converting from f32 source on the fly.
DEV v16bf load_a_f32(const float* base, int ld, int m0, int k0, int lane) {
    int r  = m0 + (lane & 15);
    int kb = k0 + ((lane >> 4) << 3);
    const float* p = base + (size_t)r * ld + kb;
    v16bf v;
#pragma unroll
    for (int j = 0; j < 8; ++j) {
        v[j]     = (__bf16)p[j];
        v[8 + j] = (__bf16)p[16 + j];
    }
    return v;
}

// B fragment (32x16 bf16, KxN). Source holds B^T row-major: srcT[n][k], ld given.
// ISA: lane = column (lane&15); lanes 0-15 hold K={0..7,16..23}, lanes 16-31 K={8..15,24..31}.
DEV v16bf load_bt_bf16(const __bf16* baseT, int ld, int n0, int k0, int lane) {
    int n  = n0 + (lane & 15);
    int kb = k0 + ((lane >> 4) << 3);
    const __bf16* p = baseT + (size_t)n * ld + kb;
    union { v16bf v; v8bf h[2]; } u;
    u.h[0] = *(const v8bf*)(p);
    u.h[1] = *(const v8bf*)(p + 16);
    return u.v;
}

DEV v8f v8f_zero() {
    v8f z = {0.f, 0.f, 0.f, 0.f, 0.f, 0.f, 0.f, 0.f};
    return z;
}

// --------------------------- async copy helpers ----------------------------
DEV void async_load_b128(void* lds_dst, const void* gsrc) {
    // LDS aperture occupies only the high 32 bits of a flat address; the low
    // 32 bits are the in-allocation LDS byte offset the async op wants.
    unsigned ldsoff = (unsigned)(size_t)lds_dst;
    unsigned long long ga = (unsigned long long)(size_t)gsrc;
    asm volatile("global_load_async_to_lds_b128 %0, %1, off"
                 :: "v"(ldsoff), "v"(ga) : "memory");
}

template <int N>
DEV void wait_async_le() {
#if __has_builtin(__builtin_amdgcn_s_wait_asynccnt)
    __builtin_amdgcn_s_wait_asynccnt(N);
#else
    asm volatile("s_wait_asynccnt %0" :: "i"(N) : "memory");
#endif
}

// ---------------------------------------------------------------------------
// K1: Fc = Wf*Fn + bf ; Ac = Wa*An + ba.  Per block: one batch, 32 hw columns.
// Activations staged to LDS transposed (f32->bf16) so B fragments are LDS-contig.
// Also writes FcbT (column-major [hw][cc]) via packed v8bf epilogue stores.
// ---------------------------------------------------------------------------
__global__ void __launch_bounds__(256)
k_conv_fa(const float* __restrict__ Fn, const float* __restrict__ An,
          const float* __restrict__ Wf, const float* __restrict__ bfv,
          const float* __restrict__ Wa, const float* __restrict__ bav,
          float* __restrict__ Fc, __bf16* __restrict__ Fcb,
          __bf16* __restrict__ Acb, __bf16* __restrict__ FcbT)
{
    __shared__ __align__(16) __bf16 Xf[NT1][C_ + 8];
    __shared__ __align__(16) __bf16 Xa[NT1][C_ + 8];
    const int b   = blockIdx.y;
    const int hw0 = blockIdx.x * NT1;
    const int t = threadIdx.x, lane = t & 31, w = t >> 5;

    const float* fn = Fn + (size_t)b * C_ * HW_;
    const float* an = An + (size_t)b * C_ * HW_;
#pragma unroll
    for (int it = 0; it < (NT1 * C_) / 256; ++it) {
        int idx = t + 256 * it;
        int c = idx >> 5;          // NT1 == 32
        int j = idx & (NT1 - 1);
        Xf[j][c] = (__bf16)fn[(size_t)c * HW_ + hw0 + j];
        Xa[j][c] = (__bf16)an[(size_t)c * HW_ + hw0 + j];
    }
    __syncthreads();

    const int m0 = w * 16;                       // wave -> one of 8 M tiles
    v8f aF[2] = {v8f_zero(), v8f_zero()};
    v8f aA[2] = {v8f_zero(), v8f_zero()};
    for (int k0 = 0; k0 < C_; k0 += 32) {
        v16bf wf = load_a_f32(Wf, C_, m0, k0, lane);
        v16bf wa = load_a_f32(Wa, C_, m0, k0, lane);
#pragma unroll
        for (int nt = 0; nt < 2; ++nt) {
            v16bf bF = load_bt_bf16(&Xf[0][0], C_ + 8, nt * 16, k0, lane);
            v16bf bA = load_bt_bf16(&Xa[0][0], C_ + 8, nt * 16, k0, lane);
            aF[nt] = wmma_bf16(wf, bF, aF[nt]);
            aA[nt] = wmma_bf16(wa, bA, aA[nt]);
        }
    }

    float*  fcO = Fc   + (size_t)b * CC_ * HW_;
    __bf16* fbO = Fcb  + (size_t)b * CC_ * HW_;
    __bf16* abO = Acb  + (size_t)b * CC_ * HW_;
    __bf16* ftO = FcbT + (size_t)b * HW_ * CC_;
    const int mo = m0 + ((lane >> 4) << 3);
#pragma unroll
    for (int nt = 0; nt < 2; ++nt) {
        int n = hw0 + nt * 16 + (lane & 15);
        v8bf pkT;
#pragma unroll
        for (int r = 0; r < 8; ++r) {
            size_t o = (size_t)(mo + r) * HW_ + n;
            float vF = aF[nt][r] + bfv[mo + r];
            float vA = aA[nt][r] + bav[mo + r];
            fcO[o] = vF;
            fbO[o] = (__bf16)vF;
            abO[o] = (__bf16)vA;
            pkT[r] = (__bf16)vF;
        }
        *(v8bf*)(ftO + (size_t)n * CC_ + mo) = pkT;   // column-major copy
    }
}

// ---------------------------------------------------------------------------
// K2: attn[b] = Fc_flat * Ac_flat^T  (128x128 per batch, K = 16384)
// Double-buffered LDS slabs (K=64) filled with async b128 copies.
// ---------------------------------------------------------------------------
constexpr int KS2 = 64;
constexpr int KP2 = KS2 + 8;

DEV void issue_slab(const __bf16* Abase, const __bf16* Bbase, int k,
                    __bf16* lA, __bf16* lB, int t)
{
#pragma unroll
    for (int it = 0; it < 5; ++it) {
        int task = t + 256 * it;              // 0..1279 ; 144 rows * 8 chunks used
        int row  = task >> 3;
        if (row > 143) row = 143;             // duplicate copies are benign
        int l8   = task & 7;                  // 16B chunk within a 64-elem row
        const __bf16* src;
        __bf16* dst;
        if (row < 16) { src = Abase + (size_t)row * HW_ + k;        dst = lA + row * KP2; }
        else          { int rb = row - 16;
                        src = Bbase + (size_t)rb * HW_ + k;         dst = lB + rb * KP2; }
        async_load_b128(dst + l8 * 8, src + l8 * 8);
    }
}

__global__ void __launch_bounds__(256)
k_attn(const __bf16* __restrict__ Fcb, const __bf16* __restrict__ Acb,
       float* __restrict__ attn)
{
    __shared__ __align__(16) __bf16 As[2][16][KP2];
    __shared__ __align__(16) __bf16 Bs[2][128][KP2];
    const int b = blockIdx.y, mi = blockIdx.x;
    const int t = threadIdx.x, lane = t & 31, w = t >> 5;
    const __bf16* Abase = Fcb + (size_t)b * CC_ * HW_ + (size_t)(mi * 16) * HW_;
    const __bf16* Bbase = Acb + (size_t)b * CC_ * HW_;   // B^T rows = Ac rows
    const int m0 = mi * 16, n0 = w * 16;

    issue_slab(Abase, Bbase, 0, &As[0][0][0], &Bs[0][0][0], t);

    v8f acc = v8f_zero();
    int buf = 0;
    for (int k0 = 0; k0 < HW_; k0 += KS2) {
        const int nk = k0 + KS2;
        if (nk < HW_) {
            issue_slab(Abase, Bbase, nk, &As[buf ^ 1][0][0], &Bs[buf ^ 1][0][0], t);
            wait_async_le<5>();               // wait for the *current* slab only
        } else {
            wait_async_le<0>();
        }
        __syncthreads();
#pragma unroll
        for (int kk = 0; kk < KS2; kk += 32) {
            v16bf a  = load_a_bf16 (&As[buf][0][0], KP2, 0,  kk, lane);
            v16bf bb = load_bt_bf16(&Bs[buf][0][0], KP2, n0, kk, lane);
            acc = wmma_bf16(a, bb, acc);
        }
        __syncthreads();                      // slab fully consumed before reuse
        buf ^= 1;
    }

    float* o = attn + (size_t)b * CC_ * CC_;
    const int mo = m0 + ((lane >> 4) << 3);
    const int n  = n0 + (lane & 15);
#pragma unroll
    for (int r = 0; r < 8; ++r) o[(size_t)(mo + r) * CC_ + n] = acc[r];
}

// ---------------------------------------------------------------------------
// K3: row softmax over attn (rows of length 128), one wave per row.
// ---------------------------------------------------------------------------
__global__ void __launch_bounds__(256)
k_softmax(const float* __restrict__ attn, __bf16* __restrict__ attnPb)
{
    const int t = threadIdx.x, lane = t & 31, w = t >> 5;
    const int row = blockIdx.x * 8 + w;                // 0 .. B*CC-1
    const float* r = attn + (size_t)row * CC_;
    float v[4];
    float mx = -3.402823e38f;
#pragma unroll
    for (int j = 0; j < 4; ++j) { v[j] = r[lane + 32 * j]; mx = fmaxf(mx, v[j]); }
#pragma unroll
    for (int off = 16; off > 0; off >>= 1) mx = fmaxf(mx, __shfl_xor(mx, off, 32));
    float s = 0.f;
#pragma unroll
    for (int j = 0; j < 4; ++j) { v[j] = __expf(v[j] - mx); s += v[j]; }
#pragma unroll
    for (int off = 16; off > 0; off >>= 1) s += __shfl_xor(s, off, 32);
    float inv = 1.0f / s;
    __bf16* o = attnPb + (size_t)row * CC_;
#pragma unroll
    for (int j = 0; j < 4; ++j) o[lane + 32 * j] = (__bf16)(v[j] * inv);
}

// ---------------------------------------------------------------------------
// K4: attended[b] = P(128x128) * Fc(128x16384).
// B fragments come straight from column-major FcbT (L2-resident), no LDS.
// ---------------------------------------------------------------------------
__global__ void __launch_bounds__(256)
k_attended(const __bf16* __restrict__ attnPb, const __bf16* __restrict__ FcbT,
           float* __restrict__ attended)
{
    const int b = blockIdx.y, n0g = blockIdx.x * 32;
    const int t = threadIdx.x, lane = t & 31, w = t >> 5;
    const __bf16* A  = attnPb + (size_t)b * CC_ * CC_;
    const __bf16* Bt = FcbT   + (size_t)b * HW_ * CC_;   // [n][d]
    const int m0 = w * 16;
    v8f acc[2] = {v8f_zero(), v8f_zero()};
    for (int k0 = 0; k0 < CC_; k0 += 32) {
        __builtin_prefetch((const void*)(Bt + (size_t)(n0g + (lane & 15)) * CC_ + k0 + 32), 0, 1);
        v16bf a = load_a_bf16(A, CC_, m0, k0, lane);
#pragma unroll
        for (int nt = 0; nt < 2; ++nt) {
            v16bf bb = load_bt_bf16(Bt, CC_, n0g + nt * 16, k0, lane);
            acc[nt] = wmma_bf16(a, bb, acc[nt]);
        }
    }
    float* o = attended + (size_t)b * CC_ * HW_;
    const int mo = m0 + ((lane >> 4) << 3);
#pragma unroll
    for (int nt = 0; nt < 2; ++nt) {
        int n = n0g + nt * 16 + (lane & 15);
#pragma unroll
        for (int r = 0; r < 8; ++r)
            o[(size_t)(mo + r) * HW_ + n] = acc[nt][r];
    }
}

// ---------------------------------------------------------------------------
// K4b: faithful transpose(1,2).reshape reinterpretation + residual:
//      Z[b,i,p] = attended[b, p&127, (i<<7)|(p>>7)] + Fc[b,i,p]
// emitted column-major: ZbT[b][p][i], packed 8-wide bf16 stores (coalesced).
// ---------------------------------------------------------------------------
__global__ void __launch_bounds__(256)
k_permute_add(const float* __restrict__ attended, const float* __restrict__ Fc,
              __bf16* __restrict__ ZbT)
{
    size_t gid = (size_t)blockIdx.x * 256 + threadIdx.x;  // B*HW*(CC/8) threads
    size_t bp  = gid >> 4;                                // 16 chunks of 8 i's
    int    ic  = (int)(gid & 15) * 8;
    int    b   = (int)(bp / HW_);
    int    p   = (int)(bp % HW_);
    const float* att = attended + (size_t)b * CC_ * HW_;
    const float* fc  = Fc       + (size_t)b * CC_ * HW_;
    const int nb = p >> 7, c = p & 127;
    v8bf pk;
#pragma unroll
    for (int q = 0; q < 8; ++q) {
        int i = ic + q;
        float v = att[(size_t)c * HW_ + ((i << 7) | nb)] + fc[(size_t)i * HW_ + p];
        pk[q] = (__bf16)v;
    }
    *(v8bf*)(ZbT + ((size_t)b * HW_ + p) * CC_ + ic) = pk;
}

// ---------------------------------------------------------------------------
// K5: Out = Wo(256x128) * Z(128x16384) + bo ; B fragments direct from ZbT.
// ---------------------------------------------------------------------------
__global__ void __launch_bounds__(256)
k_out(const __bf16* __restrict__ ZbT, const float* __restrict__ Wo,
      const float* __restrict__ bo, float* __restrict__ Out)
{
    const int b = blockIdx.y, p0 = blockIdx.x * 32;
    const int t = threadIdx.x, lane = t & 31, w = t >> 5;
    const __bf16* Bt = ZbT + (size_t)b * HW_ * CC_;      // [p][i]
    v8f acc[4] = {v8f_zero(), v8f_zero(), v8f_zero(), v8f_zero()};
    for (int k0 = 0; k0 < CC_; k0 += 32) {
        v16bf a0 = load_a_f32(Wo, CC_, (2 * w + 0) * 16, k0, lane);
        v16bf a1 = load_a_f32(Wo, CC_, (2 * w + 1) * 16, k0, lane);
#pragma unroll
        for (int nt = 0; nt < 2; ++nt) {
            v16bf bb = load_bt_bf16(Bt, CC_, p0 + nt * 16, k0, lane);
            acc[0 * 2 + nt] = wmma_bf16(a0, bb, acc[0 * 2 + nt]);
            acc[1 * 2 + nt] = wmma_bf16(a1, bb, acc[1 * 2 + nt]);
        }
    }
    float* o = Out + (size_t)b * C_ * HW_;
#pragma unroll
    for (int mt = 0; mt < 2; ++mt) {
        int mo = (2 * w + mt) * 16 + ((lane >> 4) << 3);
#pragma unroll
        for (int nt = 0; nt < 2; ++nt) {
            int n = p0 + nt * 16 + (lane & 15);
#pragma unroll
            for (int r = 0; r < 8; ++r)
                o[(size_t)(mo + r) * HW_ + n] = acc[mt * 2 + nt][r] + bo[mo + r];
        }
    }
}

// ---------------------------------------------------------------------------
// Launch
// ---------------------------------------------------------------------------
extern "C" void kernel_launch(void* const* d_in, const int* in_sizes, int n_in,
                              void* d_out, int out_size, void* d_ws, size_t ws_size,
                              hipStream_t stream)
{
    (void)in_sizes; (void)n_in; (void)out_size; (void)ws_size;
    const float* An = (const float*)d_in[0];
    const float* Fn = (const float*)d_in[1];
    const float* Wf = (const float*)d_in[2];
    const float* bf = (const float*)d_in[3];
    const float* Wa = (const float*)d_in[4];
    const float* ba = (const float*)d_in[5];
    const float* Wo = (const float*)d_in[6];
    const float* bo = (const float*)d_in[7];
    float* Out = (float*)d_out;

    char* ws = (char*)d_ws;
    float*  Fc       = (float*)ws;   ws += (size_t)B_ * CC_ * HW_ * sizeof(float);
    float*  attended = (float*)ws;   ws += (size_t)B_ * CC_ * HW_ * sizeof(float);
    __bf16* Fcb      = (__bf16*)ws;  ws += (size_t)B_ * CC_ * HW_ * sizeof(__bf16);
    __bf16* FcbT     = (__bf16*)ws;  ws += (size_t)B_ * CC_ * HW_ * sizeof(__bf16);
    __bf16* Acb      = (__bf16*)ws;  ws += (size_t)B_ * CC_ * HW_ * sizeof(__bf16);
    __bf16* ZbT      = (__bf16*)ws;  ws += (size_t)B_ * CC_ * HW_ * sizeof(__bf16);
    float*  attn     = (float*)ws;   ws += (size_t)B_ * CC_ * CC_ * sizeof(float);
    __bf16* attnPb   = (__bf16*)ws;  ws += (size_t)B_ * CC_ * CC_ * sizeof(__bf16);

    k_conv_fa    <<<dim3(HW_ / NT1, B_), 256, 0, stream>>>(Fn, An, Wf, bf, Wa, ba,
                                                           Fc, Fcb, Acb, FcbT);
    k_attn       <<<dim3(CC_ / 16, B_),  256, 0, stream>>>(Fcb, Acb, attn);
    k_softmax    <<<dim3(B_ * CC_ / 8),  256, 0, stream>>>(attn, attnPb);
    k_attended   <<<dim3(HW_ / 32, B_),  256, 0, stream>>>(attnPb, FcbT, attended);
    k_permute_add<<<dim3((B_ * HW_ * (CC_ / 8)) / 256), 256, 0, stream>>>(attended, Fc, ZbT);
    k_out        <<<dim3(HW_ / 32, B_),  256, 0, stream>>>(ZbT, Wo, bo, Out);
}
